// GraphEncoder_48275432407137
// MI455X (gfx1250) — matrix operand order repaired and checked by reading, not compiled, not run on hardware
//
#include <hip/hip_runtime.h>
#include <hip/hip_bf16.h>

#define DD 256
#define NODE_VOCAB 120
#define BOND_VOCAB 6
#define NUM_NODE_ATTR 10
#define NUM_EDGE_ATTR 5
#define SELF_LOOP_TYPE (BOND_VOCAB - 1)

typedef __attribute__((ext_vector_type(16))) __bf16 v16bf;
typedef __attribute__((ext_vector_type(8)))  float  v8f;

union FragU {
    uint4 q[2];
    v16bf v;
};

// round-to-nearest-even float -> bf16 bits
static __device__ __forceinline__ unsigned short f2bf(float f) {
    unsigned u = __float_as_uint(f);
    u = u + 0x7fffu + ((u >> 16) & 1u);
    return (unsigned short)(u >> 16);
}

// ---------------------------------------------------------------------------
// Node embedding: hbf[n,d] = bf16( sum_a node_emb[a, x[n,a], d] )
// ---------------------------------------------------------------------------
__global__ void k_embed(const int* __restrict__ x,
                        const float* __restrict__ ne,
                        unsigned short* __restrict__ hbf, int n) {
    int nid = blockIdx.x;
    int d = threadIdx.x;
    if (nid >= n) return;
    const int* xr = x + (size_t)nid * NUM_NODE_ATTR;
    float s = 0.f;
#pragma unroll
    for (int a = 0; a < NUM_NODE_ATTR; ++a) {
        int xa = xr[a];
        s += ne[((size_t)a * NODE_VOCAB + xa) * DD + d];
    }
    hbf[(size_t)nid * DD + d] = f2bf(s);
}

// ---------------------------------------------------------------------------
// fill
// ---------------------------------------------------------------------------
__global__ void k_fill(float* __restrict__ p, float v, unsigned long long n) {
    unsigned long long i = (unsigned long long)blockIdx.x * blockDim.x + threadIdx.x;
    unsigned long long stride = (unsigned long long)gridDim.x * blockDim.x;
    for (; i < n; i += stride) p[i] = v;
}

// degree count over edge sources (deg pre-initialized to 1.0 for self-loops)
__global__ void k_degcnt(const int* __restrict__ row, float* __restrict__ deg, int E) {
    int t = blockIdx.x * blockDim.x + threadIdx.x;
    if (t < E) atomicAdd(&deg[row[t]], 1.0f);
}

// deg -> deg^-0.5 in place (deg >= 1 always, self-loop included)
__global__ void k_dinv(float* __restrict__ deg, int n) {
    int t = blockIdx.x * blockDim.x + threadIdx.x;
    if (t < n) deg[t] = rsqrtf(deg[t]);
}

// norm[e] = dinv[row]*dinv[col]; self loops: dinv[n]^2
__global__ void k_norm(const float* __restrict__ dinv,
                       const int* __restrict__ row, const int* __restrict__ col,
                       float* __restrict__ norm, int E, int n) {
    int t = blockIdx.x * blockDim.x + threadIdx.x;
    if (t >= E + n) return;
    if (t < E) norm[t] = dinv[row[t]] * dinv[col[t]];
    else { float v = dinv[t - E]; norm[t] = v * v; }
}

// ---------------------------------------------------------------------------
// Convert W[l][k][n] (f32) -> Wt[l][n][k] (bf16, transposed)
// ---------------------------------------------------------------------------
__global__ void k_wcvt(const float* __restrict__ W, unsigned short* __restrict__ Wt, int total) {
    int i = blockIdx.x * blockDim.x + threadIdx.x;
    int stride = gridDim.x * blockDim.x;
    for (; i < total; i += stride) {
        int l = i / (DD * DD);
        int rem = i - l * (DD * DD);
        int k = rem / DD;
        int n = rem - k * DD;
        Wt[(size_t)l * DD * DD + (size_t)n * DD + k] = f2bf(W[i]);
    }
}

// ---------------------------------------------------------------------------
// GEMM: out[r,n] = sum_k h[r,k] * W[k,n] + bias[n]
// hbf: A in bf16 row-major; Wt: B in bf16 [n][k] layout.
// Block = 128 threads = 4 waves; block tile 16 rows x 64 cols
// (one 16x16 v_wmma_f32_16x16x32_bf16 tile per wave, 8 K-steps).
// A strip staged in LDS (row stride 264 bf16 to spread banks).
// ---------------------------------------------------------------------------
#define A_STRIDE 264

__global__ void k_gemm(const unsigned short* __restrict__ hbf,
                       const unsigned short* __restrict__ Wt,
                       const float* __restrict__ bias,
                       float* __restrict__ out, int nrows) {
    __shared__ __align__(16) unsigned short sA[16 * A_STRIDE];
    const int tid = threadIdx.x;
    const int row0 = blockIdx.x * 16;
    const bool full = (row0 + 16 <= nrows);

    // stage A (16 x 256 bf16) into LDS with uint4 copies (8 bf16 per copy)
    for (int i = tid; i < 16 * 32; i += 128) {
        int r = i >> 5;
        int c = i & 31;
        uint4 v = {0u, 0u, 0u, 0u};
        int gr = row0 + r;
        if (full || gr < nrows)
            v = *(const uint4*)(hbf + (size_t)gr * DD + c * 8);
        *(uint4*)(&sA[r * A_STRIDE + c * 8]) = v;
    }
    __syncthreads();

    const int wave = tid >> 5;
    const int lane = tid & 31;
    const int lo = lane & 15;
    const int hi = lane >> 4;       // half-wave select
    const int tileN = blockIdx.y * 64 + wave * 16;
    const int nIdx = tileN + lo;    // C/D column owned by this lane

    v8f acc;
    float bv = bias[nIdx];
#pragma unroll
    for (int i = 0; i < 8; ++i) acc[i] = bv;

    const unsigned short* wcol = Wt + (size_t)nIdx * DD;

#pragma unroll
    for (int k0 = 0; k0 < 256; k0 += 32) {
        FragU fa, fb;
        // A 16x32 bf16 layout: lane(lo)=row M; elems 0..7 -> K=k0+hi*8+0..7,
        // elems 8..15 -> K=k0+16+hi*8+0..7
        const int aBase = lo * A_STRIDE + k0 + hi * 8;
        fa.q[0] = *(const uint4*)(&sA[aBase]);
        fa.q[1] = *(const uint4*)(&sA[aBase + 16]);
        // B 32x16 bf16 layout: lane(lo)=col N; elems 0..7 -> K=k0+hi*8+0..7,
        // elems 8..15 -> K=k0+16+hi*8+0..7
        const int kb = k0 + hi * 8;
        fb.q[0] = *(const uint4*)(wcol + kb);
        fb.q[1] = *(const uint4*)(wcol + kb + 16);
        acc = __builtin_amdgcn_wmma_f32_16x16x32_bf16(
            false, fa.v, false, fb.v, (short)0, acc, false, false);
    }

    // C/D layout: VGPR i -> M = hi*8 + i, N = nIdx
    if (full) {
        float* o = out + (size_t)(row0 + hi * 8) * DD + nIdx;
#pragma unroll
        for (int i = 0; i < 8; ++i) o[(size_t)i * DD] = acc[i];
    } else {
#pragma unroll
        for (int i = 0; i < 8; ++i) {
            int r = row0 + hi * 8 + i;
            if (r < nrows) out[(size_t)r * DD + nIdx] = acc[i];
        }
    }
}

// ---------------------------------------------------------------------------
// Message + scatter-add. One block per edge (incl. self loops), 256 threads.
// ---------------------------------------------------------------------------
__global__ void k_scatter(const float* __restrict__ hl,
                          const float* __restrict__ norm,
                          const int* __restrict__ row, const int* __restrict__ col,
                          const int* __restrict__ ea,
                          const float* __restrict__ eemb,  // [5][6][256] for this layer
                          float* __restrict__ agg, int E, int n) {
    int e = blockIdx.x;
    int d = threadIdx.x;
    int r, c, a0, a1, a2, a3, a4;
    if (e < E) {
        r = row[e];
        c = col[e];
        const int* p = ea + (size_t)e * NUM_EDGE_ATTR;
        a0 = p[0]; a1 = p[1]; a2 = p[2]; a3 = p[3]; a4 = p[4];
    } else {
        r = c = e - E;
        a0 = SELF_LOOP_TYPE; a1 = a2 = a3 = a4 = 0;
    }
    float ev = eemb[(0 * BOND_VOCAB + a0) * DD + d]
             + eemb[(1 * BOND_VOCAB + a1) * DD + d]
             + eemb[(2 * BOND_VOCAB + a2) * DD + d]
             + eemb[(3 * BOND_VOCAB + a3) * DD + d]
             + eemb[(4 * BOND_VOCAB + a4) * DD + d];
    ev *= 0.2f;  // mean over 5 attrs
    float m = norm[e] * (hl[(size_t)r * DD + d] + ev);
    atomicAdd(&agg[(size_t)c * DD + d], m);
}

// ---------------------------------------------------------------------------
// LayerNorm (+ optional ReLU). One block (256 thr) per node.
// Non-last layers: write bf16 h for the next GEMM. Last layer: write f32 out.
// ---------------------------------------------------------------------------
__global__ void k_ln(const float* __restrict__ agg,
                     unsigned short* __restrict__ hbf,
                     float* __restrict__ outf,
                     const float* __restrict__ g, const float* __restrict__ bt,
                     int relu, int last, int n) {
    __shared__ float red[256];
    int nid = blockIdx.x;
    int d = threadIdx.x;
    float x = agg[(size_t)nid * DD + d];
    red[d] = x;
    __syncthreads();
    for (int s = 128; s > 0; s >>= 1) {
        if (d < s) red[d] += red[d + s];
        __syncthreads();
    }
    float mu = red[0] * (1.0f / DD);
    __syncthreads();
    float cc = x - mu;
    red[d] = cc * cc;
    __syncthreads();
    for (int s = 128; s > 0; s >>= 1) {
        if (d < s) red[d] += red[d + s];
        __syncthreads();
    }
    float var = red[0] * (1.0f / DD);
    float y = g[d] * cc * rsqrtf(var + 1e-5f) + bt[d];
    if (relu) y = fmaxf(y, 0.f);
    if (last) outf[(size_t)nid * DD + d] = y;
    else      hbf[(size_t)nid * DD + d] = f2bf(y);
}

// ---------------------------------------------------------------------------
extern "C" void kernel_launch(void* const* d_in, const int* in_sizes, int n_in,
                              void* d_out, int out_size, void* d_ws, size_t ws_size,
                              hipStream_t stream) {
    const int*   x        = (const int*)d_in[0];
    const int*   eidx     = (const int*)d_in[1];
    const int*   ea       = (const int*)d_in[2];
    const float* node_emb = (const float*)d_in[3];
    const float* edge_emb = (const float*)d_in[4];
    const float* W        = (const float*)d_in[5];
    const float* b        = (const float*)d_in[6];
    const float* gamma    = (const float*)d_in[7];
    const float* beta     = (const float*)d_in[8];

    const int N = in_sizes[0] / NUM_NODE_ATTR;
    const int E = in_sizes[1] / 2;
    const int L = in_sizes[5] / (DD * DD);
    const int* row = eidx;
    const int* col = eidx + E;

    float* agg = (float*)d_out;  // aggregation target + final output

    // workspace carve-up
    char* ws = (char*)d_ws;
    size_t off = 0;
    auto carve = [&](size_t bytes) -> void* {
        void* p = ws + off;
        off += (bytes + 255) & ~(size_t)255;
        return p;
    };
    float*          hl  = (float*)carve((size_t)N * DD * sizeof(float));
    unsigned short* hbf = (unsigned short*)carve((size_t)N * DD * sizeof(unsigned short));
    unsigned short* Wt  = (unsigned short*)carve((size_t)L * DD * DD * sizeof(unsigned short));
    float*          nrm = (float*)carve((size_t)(E + N) * sizeof(float));
    float*          deg = (float*)carve((size_t)N * sizeof(float));

    // one-time prep
    k_wcvt<<<512, 256, 0, stream>>>(W, Wt, L * DD * DD);
    k_embed<<<N, DD, 0, stream>>>(x, node_emb, hbf, N);
    k_fill<<<1024, 256, 0, stream>>>(deg, 1.0f, (unsigned long long)N);
    k_degcnt<<<(E + 255) / 256, 256, 0, stream>>>(row, deg, E);
    k_dinv<<<(N + 255) / 256, 256, 0, stream>>>(deg, N);
    k_norm<<<(E + N + 255) / 256, 256, 0, stream>>>(deg, row, col, nrm, E, N);

    for (int l = 0; l < L; ++l) {
        dim3 g((N + 15) / 16, DD / 64);
        k_gemm<<<g, 128, 0, stream>>>(hbf, Wt + (size_t)l * DD * DD, b + (size_t)l * DD, hl, N);
        k_fill<<<4096, 256, 0, stream>>>(agg, 0.f, (unsigned long long)N * DD);
        k_scatter<<<E + N, DD, 0, stream>>>(hl, nrm, row, col, ea,
                                            edge_emb + (size_t)l * NUM_EDGE_ATTR * BOND_VOCAB * DD,
                                            agg, E, N);
        k_ln<<<N, DD, 0, stream>>>(agg, hbf, agg,
                                   gamma + (size_t)l * DD, beta + (size_t)l * DD,
                                   (l < L - 1) ? 1 : 0, (l == L - 1) ? 1 : 0, N);
    }
}
